// DeformableAttentionModule_3341484556406
// MI455X (gfx1250) — matrix operand order repaired and checked by reading, not compiled, not run on hardware
//
#include <hip/hip_runtime.h>

typedef __attribute__((ext_vector_type(16))) _Float16 v16h;
typedef __attribute__((ext_vector_type(8)))  _Float16 v8h;
typedef __attribute__((ext_vector_type(8)))  float    v8f;

#define BATCH   8
#define CDIM    256
#define HW      128
#define NQ      1024      // (128/4)^2
#define NPTS    8         // NHEAD sampling points
#define GRID    32        // sqrt(NQ)

// ---------------------------------------------------------------------------
// CDNA5 async global->LDS copy (ASYNCcnt path, gfx1250).
// lds_off is the byte offset within the workgroup's LDS allocation
// (= low 32 bits of the generic shared-aperture address per ISA 10.2).
// ---------------------------------------------------------------------------
__device__ __forceinline__ void async_copy_b128(void* lds_ptr, const void* gptr) {
    unsigned int lds_off = (unsigned int)(unsigned long long)lds_ptr;
    asm volatile("global_load_async_to_lds_b128 %0, %1, off"
                 :: "v"(lds_off), "v"(gptr)
                 : "memory");
}
__device__ __forceinline__ void async_wait0() {
    asm volatile("s_wait_asynccnt 0x0" ::: "memory");
}

// ---------------------------------------------------------------------------
// Bilinear sample from one channel plane (align_corners, clamped like ref)
// ---------------------------------------------------------------------------
__device__ __forceinline__ float bilin(const float* __restrict__ plane, float px, float py) {
    float x0 = floorf(px); x0 = fminf(fmaxf(x0, 0.f), 126.f);
    float y0 = floorf(py); y0 = fminf(fmaxf(y0, 0.f), 126.f);
    float wx = fminf(fmaxf(px - x0, 0.f), 1.f);
    float wy = fminf(fmaxf(py - y0, 0.f), 1.f);
    int xi = (int)x0, yi = (int)y0;
    const float* p0 = plane + yi * HW + xi;
    float v00 = p0[0], v01 = p0[1], v10 = p0[HW], v11 = p0[HW + 1];
    return v00 * (1.f - wx) * (1.f - wy) + v01 * wx * (1.f - wy)
         + v10 * (1.f - wx) * wy        + v11 * wx * wy;
}

// ---------------------------------------------------------------------------
// K0: transpose + f32->f16 the three projection weights: Wt[k][n] = W[n][k]
// grid (256, 3), block 256
// ---------------------------------------------------------------------------
__global__ void k_prep_w(const float* __restrict__ Wq, const float* __restrict__ Wk,
                         const float* __restrict__ Wv,
                         _Float16* __restrict__ Tq, _Float16* __restrict__ Tk,
                         _Float16* __restrict__ Tv) {
    int k = blockIdx.x, n = threadIdx.x;
    const float* src; _Float16* dst;
    if (blockIdx.y == 0)      { src = Wq; dst = Tq; }
    else if (blockIdx.y == 1) { src = Wk; dst = Tk; }
    else                      { src = Wv; dst = Tv; }
    dst[k * CDIM + n] = (_Float16)src[n * CDIM + k];
}

// ---------------------------------------------------------------------------
// K1: sample feature map at reference grid points -> q_feat (f32 and f16)
// grid (NQ, B), block 256 (one thread per channel)
// ---------------------------------------------------------------------------
__global__ void k_sample_q(const float* __restrict__ fm, float* __restrict__ qf32,
                           _Float16* __restrict__ qf16) {
    int n = blockIdx.x, b = blockIdx.y, c = threadIdx.x;
    int ix = n & (GRID - 1), iy = n >> 5;
    float xn = -1.f + (2.f / 31.f) * (float)ix;
    float yn = -1.f + (2.f / 31.f) * (float)iy;
    float px = (xn + 1.f) * 0.5f * 127.f;
    float py = (yn + 1.f) * 0.5f * 127.f;
    const float* plane = fm + ((size_t)(b * CDIM + c)) * (HW * HW);
    float v = bilin(plane, px, py);
    size_t o = ((size_t)(b * NQ + n)) * CDIM + c;
    qf32[o] = v;
    qf16[o] = (_Float16)v;
}

// ---------------------------------------------------------------------------
// K2: offset MLP: h = gelu(LN(qf @ W_o1 + b_o1)); off = (h @ W_o2 + b_o2)*4
//     pts = ref + off * [2/127, 2/127]   (normalized coords)
// grid 8192, block 64
// ---------------------------------------------------------------------------
__global__ void k_offsets(const float* __restrict__ qf32,
                          const float* __restrict__ Wo1, const float* __restrict__ bo1,
                          const float* __restrict__ lng, const float* __restrict__ lnb,
                          const float* __restrict__ Wo2, const float* __restrict__ bo2,
                          float* __restrict__ pts) {
    __shared__ float qs[CDIM];
    __shared__ float hs[64];
    __shared__ float red[64], red2[64];
    int bn = blockIdx.x;
    int t  = threadIdx.x;   // 0..63
    const float* q = qf32 + (size_t)bn * CDIM;
    qs[t] = q[t]; qs[t + 64] = q[t + 64]; qs[t + 128] = q[t + 128]; qs[t + 192] = q[t + 192];
    __syncthreads();

    float acc = bo1[t];
    #pragma unroll 4
    for (int c = 0; c < CDIM; ++c) acc += qs[c] * Wo1[c * 64 + t];

    red[t] = acc; red2[t] = acc * acc;
    __syncthreads();
    for (int s = 32; s > 0; s >>= 1) {
        if (t < s) { red[t] += red[t + s]; red2[t] += red2[t + s]; }
        __syncthreads();
    }
    float mu  = red[0] * (1.f / 64.f);
    float var = red2[0] * (1.f / 64.f) - mu * mu;
    float xh  = (acc - mu) * rsqrtf(var + 1e-5f) * lng[t] + lnb[t];
    // tanh-approx GELU (jax.nn.gelu default)
    float g = 0.5f * xh * (1.f + tanhf(0.7978845608028654f * (xh + 0.044715f * xh * xh * xh)));
    hs[t] = g;
    __syncthreads();

    if (t < 16) {
        float o = bo2[t];
        #pragma unroll
        for (int i = 0; i < 64; ++i) o += hs[i] * Wo2[i * 16 + t];
        o *= 4.0f;                               // OFFSET_SCALE
        int n  = bn & (NQ - 1);
        int ix = n & (GRID - 1), iy = n >> 5;
        float refv = (t & 1) ? (-1.f + (2.f / 31.f) * (float)iy)
                             : (-1.f + (2.f / 31.f) * (float)ix);
        pts[(size_t)bn * 16 + t] = refv + o * (2.f / 127.f);
    }
}

// ---------------------------------------------------------------------------
// K3: sample feature map at the 8 deformed points -> sampled f16 [B*NQ*8, 256]
// grid (NQ, B), block 256
// ---------------------------------------------------------------------------
__global__ void k_sample_kv(const float* __restrict__ fm, const float* __restrict__ pts,
                            _Float16* __restrict__ sampled) {
    __shared__ float sp[16];
    int n = blockIdx.x, b = blockIdx.y, c = threadIdx.x;
    size_t bn = (size_t)b * NQ + n;
    if (c < 16) sp[c] = pts[bn * 16 + c];
    __syncthreads();
    const float* plane = fm + ((size_t)(b * CDIM + c)) * (HW * HW);
    #pragma unroll
    for (int p = 0; p < NPTS; ++p) {
        float px = (sp[2 * p]     + 1.f) * 0.5f * 127.f;
        float py = (sp[2 * p + 1] + 1.f) * 0.5f * 127.f;
        float v = bilin(plane, px, py);
        sampled[(bn * NPTS + p) * CDIM + c] = (_Float16)v;
    }
}

// ---------------------------------------------------------------------------
// WMMA fragment loaders.
//  A (16x32 f16, row-major src, K contiguous): lane<16 holds K{0..7,16..23},
//  lane>=16 holds K{8..15,24..31} of row M=lane&15.
//  B (32x16 f16): lane L holds column N=L&15, same K order as A (Wt = W^T
//  row-major [K][N] staged in LDS, gathered per element).
// ---------------------------------------------------------------------------
__device__ __forceinline__ v16h load_a_frag(const _Float16* __restrict__ rowp, int lane) {
    const _Float16* ap = rowp + ((lane & 16) ? 8 : 0);
    v8h lo = *(const v8h*)(ap);
    v8h hi = *(const v8h*)(ap + 16);
    return __builtin_shufflevector(lo, hi, 0, 1, 2, 3, 4, 5, 6, 7,
                                           8, 9, 10, 11, 12, 13, 14, 15);
}

// Stage one 256(K) x 16(N) f16 weight panel into LDS with async-to-LDS copies.
// panel row k lives at &Bs[k*16], 32B aligned. 512 x 16B chunks, 256 threads.
__device__ __forceinline__ void stage_b_panel(_Float16* Bs, const _Float16* __restrict__ Bt,
                                              int nt, int t) {
    const _Float16* gB = Bt + nt * 16;
    #pragma unroll
    for (int i = t; i < 512; i += 256) {
        int k = i >> 1, h = i & 1;
        async_copy_b128(&Bs[(size_t)i * 8], gB + (size_t)k * CDIM + h * 8);
    }
    async_wait0();
}

// ---------------------------------------------------------------------------
// K4: queries = q_feat16 @ W_q^T  (M=8192, N=K=256), f32 out.
// Block = 8 waves computing a 128(M)x16(N) panel; B panel shared via LDS.
// grid = (M/128)*16 blocks.
// ---------------------------------------------------------------------------
__global__ void k_gemm_q(const _Float16* __restrict__ A, const _Float16* __restrict__ Bt,
                         float* __restrict__ D) {
    __shared__ _Float16 Bs[256 * 16];            // 8 KB
    int t    = threadIdx.x;
    int lane = t & 31;
    int nt   = blockIdx.x & 15;
    int mt   = (blockIdx.x >> 4) * 8 + (t >> 5);

    stage_b_panel(Bs, Bt, nt, t);
    __syncthreads();

    int row = mt * 16 + (lane & 15);
    const _Float16* arow = A + (size_t)row * CDIM;
    __builtin_prefetch(arow, 0, 3);              // global_prefetch_b8
    v8f acc = {0.f, 0.f, 0.f, 0.f, 0.f, 0.f, 0.f, 0.f};
    #pragma unroll
    for (int kt = 0; kt < 8; ++kt) {
        v16h a = load_a_frag(arow + kt * 32, lane);
        v16h b;
        #pragma unroll
        for (int s = 0; s < 16; ++s)
            b[s] = Bs[(kt * 32 + ((s & 8) << 1) + ((lane & 16) >> 1) + (s & 7)) * 16 + (lane & 15)];
        acc = __builtin_amdgcn_wmma_f32_16x16x32_f16(false, a, false, b,
                                                     (short)0, acc, false, false);
    }
    int col   = nt * 16 + (lane & 15);
    int rbase = mt * 16 + ((lane & 16) ? 8 : 0);
    #pragma unroll
    for (int v = 0; v < 8; ++v)
        D[(size_t)(rbase + v) * CDIM + col] = acc[v];
}

// ---------------------------------------------------------------------------
// K5: keys/values = sampled @ {W_k^T, W_v^T}  (M=65536, N=K=256), f16 out.
// Shares the A fragment between the two WMMAs; both weight panels in LDS.
// grid = (M/128)*16 blocks.
// ---------------------------------------------------------------------------
__global__ void k_gemm_kv(const _Float16* __restrict__ A, const _Float16* __restrict__ Btk,
                          const _Float16* __restrict__ Btv,
                          _Float16* __restrict__ Ko, _Float16* __restrict__ Vo) {
    __shared__ _Float16 Bks[256 * 16];           // 8 KB
    __shared__ _Float16 Bvs[256 * 16];           // 8 KB
    int t    = threadIdx.x;
    int lane = t & 31;
    int nt   = blockIdx.x & 15;
    int mt   = (blockIdx.x >> 4) * 8 + (t >> 5);

    stage_b_panel(Bks, Btk, nt, t);
    stage_b_panel(Bvs, Btv, nt, t);
    __syncthreads();

    int row = mt * 16 + (lane & 15);
    const _Float16* arow = A + (size_t)row * CDIM;
    __builtin_prefetch(arow, 0, 3);              // global_prefetch_b8
    v8f acck = {0.f, 0.f, 0.f, 0.f, 0.f, 0.f, 0.f, 0.f};
    v8f accv = {0.f, 0.f, 0.f, 0.f, 0.f, 0.f, 0.f, 0.f};
    #pragma unroll
    for (int kt = 0; kt < 8; ++kt) {
        v16h a  = load_a_frag(arow + kt * 32, lane);
        v16h bk, bv;
        #pragma unroll
        for (int s = 0; s < 16; ++s) {
            int bi = (kt * 32 + ((s & 8) << 1) + ((lane & 16) >> 1) + (s & 7)) * 16 + (lane & 15);
            bk[s] = Bks[bi];
            bv[s] = Bvs[bi];
        }
        acck = __builtin_amdgcn_wmma_f32_16x16x32_f16(false, a, false, bk,
                                                      (short)0, acck, false, false);
        accv = __builtin_amdgcn_wmma_f32_16x16x32_f16(false, a, false, bv,
                                                      (short)0, accv, false, false);
    }
    int col   = nt * 16 + (lane & 15);
    int rbase = mt * 16 + ((lane & 16) ? 8 : 0);
    #pragma unroll
    for (int v = 0; v < 8; ++v) {
        Ko[(size_t)(rbase + v) * CDIM + col] = (_Float16)acck[v];
        Vo[(size_t)(rbase + v) * CDIM + col] = (_Float16)accv[v];
    }
}

// ---------------------------------------------------------------------------
// K6: per-query 8-head attention over P=8 points.
// grid (NQ, B), block 256
// ---------------------------------------------------------------------------
__global__ void k_attn(const float* __restrict__ Q, const _Float16* __restrict__ Kk,
                       const _Float16* __restrict__ Vv, float* __restrict__ out) {
    __shared__ float qs[CDIM];
    __shared__ float ks[NPTS * CDIM];
    __shared__ float vs[NPTS * CDIM];
    __shared__ float attn[64];
    int n = blockIdx.x, b = blockIdx.y, c = threadIdx.x;
    size_t bn = (size_t)b * NQ + n;
    qs[c] = Q[bn * CDIM + c];
    #pragma unroll
    for (int p = 0; p < NPTS; ++p) {
        ks[p * CDIM + c] = (float)Kk[(bn * NPTS + p) * CDIM + c];
        vs[p * CDIM + c] = (float)Vv[(bn * NPTS + p) * CDIM + c];
    }
    __syncthreads();
    if (c < 64) {
        int h = c >> 3, p = c & 7;
        const float* qh = qs + h * 32;
        const float* kh = ks + p * CDIM + h * 32;
        float s = 0.f;
        #pragma unroll
        for (int i = 0; i < 32; ++i) s += qh[i] * kh[i];
        attn[c] = s * 0.17677669529663687f;   // 1/sqrt(32)
    }
    __syncthreads();
    if (c < 8) {
        float m = -1e30f;
        #pragma unroll
        for (int p = 0; p < 8; ++p) m = fmaxf(m, attn[c * 8 + p]);
        float e[8]; float sum = 0.f;
        #pragma unroll
        for (int p = 0; p < 8; ++p) { e[p] = __expf(attn[c * 8 + p] - m); sum += e[p]; }
        float inv = 1.f / sum;
        #pragma unroll
        for (int p = 0; p < 8; ++p) attn[c * 8 + p] = e[p] * inv;
    }
    __syncthreads();
    int h = c >> 5;
    float o = 0.f;
    #pragma unroll
    for (int p = 0; p < 8; ++p) o += attn[h * 8 + p] * vs[p * CDIM + c];
    out[bn * CDIM + c] = o;
}

// ---------------------------------------------------------------------------
extern "C" void kernel_launch(void* const* d_in, const int* in_sizes, int n_in,
                              void* d_out, int out_size, void* d_ws, size_t ws_size,
                              hipStream_t stream) {
    const float* fm  = (const float*)d_in[0];
    const float* Wq  = (const float*)d_in[1];
    const float* Wk  = (const float*)d_in[2];
    const float* Wv  = (const float*)d_in[3];
    const float* Wo1 = (const float*)d_in[4];
    const float* bo1 = (const float*)d_in[5];
    const float* lng = (const float*)d_in[6];
    const float* lnb = (const float*)d_in[7];
    const float* Wo2 = (const float*)d_in[8];
    const float* bo2 = (const float*)d_in[9];
    float* out = (float*)d_out;

    const size_t BN  = (size_t)BATCH * NQ;        // 8192
    const size_t BNP = BN * NPTS;                 // 65536

    char* w = (char*)d_ws;
    auto carve = [&](size_t bytes) {
        void* p = (void*)w;
        w += (bytes + 255) & ~(size_t)255;
        return p;
    };
    _Float16* Tq      = (_Float16*)carve((size_t)CDIM * CDIM * 2);
    _Float16* Tk      = (_Float16*)carve((size_t)CDIM * CDIM * 2);
    _Float16* Tv      = (_Float16*)carve((size_t)CDIM * CDIM * 2);
    float*    qf32    = (float*)   carve(BN * CDIM * 4);
    _Float16* qf16    = (_Float16*)carve(BN * CDIM * 2);
    float*    pts     = (float*)   carve(BN * 16 * 4);
    _Float16* sampled = (_Float16*)carve(BNP * CDIM * 2);
    _Float16* keys    = (_Float16*)carve(BNP * CDIM * 2);
    _Float16* vals    = (_Float16*)carve(BNP * CDIM * 2);
    float*    queries = (float*)   carve(BN * CDIM * 4);

    k_prep_w   <<<dim3(CDIM, 3), CDIM, 0, stream>>>(Wq, Wk, Wv, Tq, Tk, Tv);
    k_sample_q <<<dim3(NQ, BATCH), CDIM, 0, stream>>>(fm, qf32, qf16);
    k_offsets  <<<(unsigned)BN, 64, 0, stream>>>(qf32, Wo1, bo1, lng, lnb, Wo2, bo2, pts);
    k_sample_kv<<<dim3(NQ, BATCH), CDIM, 0, stream>>>(fm, pts, sampled);
    k_gemm_q   <<<(unsigned)((BN / 128) * 16), 256, 0, stream>>>(qf16, Tq, queries);      // 1024 blocks
    k_gemm_kv  <<<(unsigned)((BNP / 128) * 16), 256, 0, stream>>>(sampled, Tk, Tv, keys, vals); // 8192 blocks
    k_attn     <<<dim3(NQ, BATCH), CDIM, 0, stream>>>(queries, keys, vals, out);
}
